// WindowAttention_36163624632921
// MI455X (gfx1250) — matrix operand order repaired and checked by reading, not compiled, not run on hardware
//
#include <hip/hip_runtime.h>
#include <hip/hip_bf16.h>

typedef __attribute__((ext_vector_type(16))) _Float16 v16h;
typedef __attribute__((ext_vector_type(8)))  _Float16 v8h;
typedef __attribute__((ext_vector_type(8)))  float    v8f;

#define HEADS   8
#define HD      32
#define NWIN    49
#define DIMC    256
#define BWIN    8192

// ---------------------------------------------------------------------------
// WMMA fragment layouts (CDNA5 ISA §7.12.2, wave32), stored
// pre-swizzled in LDS so each lane's 16 halves are contiguous (2x ds_*_b128):
//   A-frag slot  = [strip][lane][j] ; j -> 16 consecutive k, k = kb+j
//                  with kb=0 (lanes 0-15) / kb=16 (lanes 16-31), m = strip*16+(lane&15)
//   B-frag slot  = [sub][lane][j]   ; j -> 16 consecutive k of column n=sub*16+(lane&15),
//                  k offset kb=0 (lanes 0-15) / 16 (lanes 16-31)
// ---------------------------------------------------------------------------
__device__ inline v8h cvt8(float4 a, float4 b) {
    v8h h;
    h[0] = (_Float16)a.x; h[1] = (_Float16)a.y; h[2] = (_Float16)a.z; h[3] = (_Float16)a.w;
    h[4] = (_Float16)b.x; h[5] = (_Float16)b.y; h[6] = (_Float16)b.z; h[7] = (_Float16)b.w;
    return h;
}

// ---------------------------------------------------------------------------
// Kernel 1: qkv = x @ W_qkv^T + b_qkv  ([401408,256] x [256,768])
// ---------------------------------------------------------------------------
__global__ __launch_bounds__(128)
void qkv_gemm_kernel(const float* __restrict__ x, const float* __restrict__ Wqkv,
                     const float* __restrict__ bqkv, _Float16* __restrict__ qkv_ws) {
    __shared__ __align__(32) _Float16 As[128 * 16];  // 4 strips x 32 lanes x 16
    __shared__ __align__(32) _Float16 Bs[128 * 16];  // 4 subtiles x 32 lanes x 16
    const int t = threadIdx.x, lane = t & 31, wave = t >> 5;
    const long m_base = (long)blockIdx.x * 64;
    const int  n_base = blockIdx.y * 64;

    const int ls   = t & 31;
    const int a_m  = (t >> 5) * 16 + (ls & 15);      // row handled for A staging
    const int a_kb = (ls < 16) ? 0 : 16;
    const int b_n  = (t >> 5) * 16 + (ls & 15);      // column handled for B staging
    const int b_kb = (ls < 16) ? 0 : 16;
    const float*  xrow = &x[(m_base + a_m) * DIMC];
    const float*  wrow = &Wqkv[(size_t)(n_base + b_n) * DIMC];

    v8f acc[4] = {};
    for (int k0 = 0; k0 < DIMC; k0 += 32) {
        __syncthreads();
        // A: two runs of 8 consecutive floats -> swizzled fragment slot
        {
            const float4* s0 = (const float4*)&xrow[k0 + a_kb];
            const float4* s1 = (const float4*)&xrow[k0 + a_kb + 8];
            *(v8h*)&As[t * 16]     = cvt8(s0[0], s0[1]);
            *(v8h*)&As[t * 16 + 8] = cvt8(s1[0], s1[1]);
        }
        // B = W^T: 16 consecutive floats of one weight row
        {
            const float4* s = (const float4*)&wrow[k0 + b_kb];
            *(v8h*)&Bs[t * 16]     = cvt8(s[0], s[1]);
            *(v8h*)&Bs[t * 16 + 8] = cvt8(s[2], s[3]);
        }
        if (k0 + 32 < DIMC)   // global_prefetch_b8 hint for next A tile
            __builtin_prefetch(&xrow[k0 + 32 + a_kb], 0, 1);
        __syncthreads();

        v16h a = *(const v16h*)&As[(wave * 32 + lane) * 16];
#pragma unroll
        for (int s = 0; s < 4; ++s) {
            v16h b = *(const v16h*)&Bs[(s * 32 + lane) * 16];
            acc[s] = __builtin_amdgcn_wmma_f32_16x16x32_f16(false, a, false, b,
                                                            (short)0, acc[s], false, false);
        }
    }

    // Fused bias + scatter into per-head q/k/v layout (f16)
#pragma unroll
    for (int s = 0; s < 4; ++s) {
        int   n    = n_base + s * 16 + (lane & 15);
        float bias = bqkv[n];
        int which = n >> 8;          // 0=q 1=k 2=v
        int d     = n & 255;
        int h     = d >> 5, hd = d & 31;
#pragma unroll
        for (int r = 0; r < 8; ++r) {
            long m   = m_base + wave * 16 + r + ((lane >> 4) << 3);
            int  b   = (int)(m / NWIN), nidx = (int)(m % NWIN);
            size_t idx = ((((size_t)which * BWIN + b) * HEADS + h) * NWIN + nidx) * HD + hd;
            qkv_ws[idx] = (_Float16)(acc[s][r] + bias);
        }
    }
}

// ---------------------------------------------------------------------------
// Kernel 2: per-(window, head) attention. 128 threads; wave w owns query
// rows [16w,16w+16). 49 padded to 64 in q-rows / k-cols.
// ---------------------------------------------------------------------------
__global__ __launch_bounds__(128)
void attn_kernel(const _Float16* __restrict__ qkv_ws, const float* __restrict__ mask,
                 _Float16* __restrict__ attn_ws) {
    __shared__ __align__(32) _Float16 Qs[128 * 16];   // swizzled A-frag layout
    __shared__ __align__(32) _Float16 Ks[64 * 32];    // row-major [kv][hd]
    __shared__ __align__(32) _Float16 Vt[32 * 64];    // transposed  [hd][kv]
    __shared__ __align__(32) float    Ss[64 * 64];    // fp32 scores [qrow][kcol]
    __shared__ __align__(32) _Float16 Ps[8 * 32 * 16];// swizzled A-frags: [strip][kc][lane][j]

    const int t = threadIdx.x, lane = t & 31, wave = t >> 5;
    const int bh = blockIdx.x;
    const int b = bh >> 3, h = bh & 7;
    const float scale = 0.17677669529663687f;   // 1/sqrt(32)

    const size_t base_q = (((size_t)0 * BWIN + b) * HEADS + h) * NWIN * HD;
    const size_t base_k = (((size_t)1 * BWIN + b) * HEADS + h) * NWIN * HD;
    const size_t base_v = (((size_t)2 * BWIN + b) * HEADS + h) * NWIN * HD;

    // ---- stage Q (swizzled), K (row-major), V (transposed) ----------------
    {   // Q: thread -> fragment slot, two contiguous 8-half runs
        int r  = (t >> 5) * 16 + (t & 15);
        int kb = ((t & 31) < 16) ? 0 : 16;
        v8h h0 = {}, h1 = {};
        if (r < NWIN) {
            h0 = *(const v8h*)&qkv_ws[base_q + r * HD + kb];
            h1 = *(const v8h*)&qkv_ws[base_q + r * HD + kb + 8];
        }
        *(v8h*)&Qs[t * 16]     = h0;
        *(v8h*)&Qs[t * 16 + 8] = h1;
    }
    {   // K: two threads per row, contiguous copy
        int row = t >> 1, half = (t & 1) * 16;
        v8h h0 = {}, h1 = {};
        if (row < NWIN) {
            h0 = *(const v8h*)&qkv_ws[base_k + row * HD + half];
            h1 = *(const v8h*)&qkv_ws[base_k + row * HD + half + 8];
        }
        *(v8h*)&Ks[row * 32 + half]     = h0;
        *(v8h*)&Ks[row * 32 + half + 8] = h1;
    }
    {   // V: contiguous global read, transpose-scatter into Vt[hd][kv]
        int rv = t >> 1, seg = (t & 1) * 16;
        v8h h0 = {}, h1 = {};
        if (rv < NWIN) {
            h0 = *(const v8h*)&qkv_ws[base_v + rv * HD + seg];
            h1 = *(const v8h*)&qkv_ws[base_v + rv * HD + seg + 8];
        }
#pragma unroll
        for (int e = 0; e < 8; ++e) {
            Vt[(seg + e) * 64 + rv]     = h0[e];
            Vt[(seg + 8 + e) * 64 + rv] = h1[e];
        }
    }
    __syncthreads();

    // ---- scores = q @ k^T (K=32, one WMMA per 16x16 tile) -----------------
    {
        v16h a = *(const v16h*)&Qs[(wave * 32 + lane) * 16];
#pragma unroll
        for (int ct = 0; ct < 4; ++ct) {
            int krow = ct * 16 + (lane & 15);
            int kb   = (lane < 16) ? 0 : 16;
            v16h bfr = *(const v16h*)&Ks[krow * 32 + kb];   // K^T column = K row
            v8f sc = {};
            sc = __builtin_amdgcn_wmma_f32_16x16x32_f16(false, a, false, bfr,
                                                        (short)0, sc, false, false);
#pragma unroll
            for (int r = 0; r < 8; ++r) {
                int mm = wave * 16 + r + ((lane >> 4) << 3);
                Ss[mm * 64 + ct * 16 + (lane & 15)] = sc[r];
            }
        }
    }
    __syncthreads();

    // ---- fp32 softmax, one row per lane (lanes 0..15 of each wave) --------
    if (lane < 16) {
        int qrow = wave * 16 + lane;
        int m    = qrow & 15;
        const float* mrow = mask + ((size_t)b * NWIN + (qrow < NWIN ? qrow : 0)) * NWIN;
        float mx = -3.0e38f;
        for (int c = 0; c < NWIN; ++c) {
            float v = Ss[qrow * 64 + c] * scale + mrow[c];
            Ss[qrow * 64 + c] = v;
            mx = fmaxf(mx, v);
        }
        float sum = 0.f;
        for (int c = 0; c < NWIN; ++c) {
            float e = __expf(Ss[qrow * 64 + c] - mx);
            Ss[qrow * 64 + c] = e;
            sum += e;
        }
        float inv = 1.f / sum;
        // write probabilities straight into swizzled A-fragment order, 8 halves/store
#pragma unroll
        for (int kc = 0; kc < 2; ++kc) {
#pragma unroll
            for (int g = 0; g < 4; ++g) {
                v8h pv;
#pragma unroll
                for (int e = 0; e < 8; ++e) {
                    int c = kc * 32 + g * 8 + e;
                    pv[e] = (c < NWIN) ? (_Float16)(Ss[qrow * 64 + c] * inv)
                                       : (_Float16)0.f;
                }
                int lane_slot = m + ((g >> 1) << 4);
                int jbase     = (g & 1) * 8;
                *(v8h*)&Ps[(((qrow >> 4) * 2 + kc) * 32 + lane_slot) * 16 + jbase] = pv;
            }
        }
    }
    __syncthreads();

    // ---- out = P(16x64) @ V(64x32): 2 K-chunks x 2 N-tiles ----------------
    v8f o[2] = {};
#pragma unroll
    for (int kc = 0; kc < 2; ++kc) {
        v16h pa = *(const v16h*)&Ps[((wave * 2 + kc) * 32 + lane) * 16];
#pragma unroll
        for (int nt = 0; nt < 2; ++nt) {
            int n  = nt * 16 + (lane & 15);
            int kb = (lane < 16) ? 0 : 16;
            v16h bv = *(const v16h*)&Vt[n * 64 + kc * 32 + kb];
            o[nt] = __builtin_amdgcn_wmma_f32_16x16x32_f16(false, pa, false, bv,
                                                           (short)0, o[nt], false, false);
        }
    }
#pragma unroll
    for (int nt = 0; nt < 2; ++nt) {
        int hd = nt * 16 + (lane & 15);
#pragma unroll
        for (int r = 0; r < 8; ++r) {
            int qrow = wave * 16 + r + ((lane >> 4) << 3);
            if (qrow < NWIN) {
                size_t idx = ((size_t)b * NWIN + qrow) * DIMC + h * HD + hd;
                attn_ws[idx] = (_Float16)o[nt][r];
            }
        }
    }
}

// ---------------------------------------------------------------------------
// Kernel 3: out = attn @ W_proj^T + b_proj  ([401408,256] x [256,256]), fp32
// ---------------------------------------------------------------------------
__global__ __launch_bounds__(128)
void proj_gemm_kernel(const _Float16* __restrict__ A, const float* __restrict__ Wp,
                      const float* __restrict__ bp, float* __restrict__ out) {
    __shared__ __align__(32) _Float16 As[128 * 16];
    __shared__ __align__(32) _Float16 Bs[128 * 16];
    const int t = threadIdx.x, lane = t & 31, wave = t >> 5;
    const long m_base = (long)blockIdx.x * 64;
    const int  n_base = blockIdx.y * 64;

    const int ls   = t & 31;
    const int a_m  = (t >> 5) * 16 + (ls & 15);
    const int a_kb = (ls < 16) ? 0 : 16;
    const int b_n  = (t >> 5) * 16 + (ls & 15);
    const int b_kb = (ls < 16) ? 0 : 16;
    const _Float16* arow = &A[(m_base + a_m) * DIMC];
    const float*    wrow = &Wp[(size_t)(n_base + b_n) * DIMC];

    v8f acc[4] = {};
    for (int k0 = 0; k0 < DIMC; k0 += 32) {
        __syncthreads();
        {   // A already f16: two contiguous 16B global loads into fragment slot
            *(v8h*)&As[t * 16]     = *(const v8h*)&arow[k0 + a_kb];
            *(v8h*)&As[t * 16 + 8] = *(const v8h*)&arow[k0 + a_kb + 8];
        }
        {
            const float4* s = (const float4*)&wrow[k0 + b_kb];
            *(v8h*)&Bs[t * 16]     = cvt8(s[0], s[1]);
            *(v8h*)&Bs[t * 16 + 8] = cvt8(s[2], s[3]);
        }
        if (k0 + 32 < DIMC)
            __builtin_prefetch(&arow[k0 + 32 + a_kb], 0, 1);
        __syncthreads();

        v16h a = *(const v16h*)&As[(wave * 32 + lane) * 16];
#pragma unroll
        for (int s = 0; s < 4; ++s) {
            v16h b = *(const v16h*)&Bs[(s * 32 + lane) * 16];
            acc[s] = __builtin_amdgcn_wmma_f32_16x16x32_f16(false, a, false, b,
                                                            (short)0, acc[s], false, false);
        }
    }
#pragma unroll
    for (int s = 0; s < 4; ++s) {
        int   n    = n_base + s * 16 + (lane & 15);
        float bias = bp[n];
#pragma unroll
        for (int r = 0; r < 8; ++r) {
            long m = m_base + wave * 16 + r + ((lane >> 4) << 3);
            out[(size_t)m * DIMC + n] = acc[s][r] + bias;
        }
    }
}

// ---------------------------------------------------------------------------
extern "C" void kernel_launch(void* const* d_in, const int* in_sizes, int n_in,
                              void* d_out, int out_size, void* d_ws, size_t ws_size,
                              hipStream_t stream) {
    const float* x     = (const float*)d_in[0];
    const float* mask  = (const float*)d_in[1];
    const float* Wqkv  = (const float*)d_in[2];
    const float* bqkv  = (const float*)d_in[3];
    const float* Wproj = (const float*)d_in[4];
    const float* bproj = (const float*)d_in[5];
    float* out = (float*)d_out;

    const size_t tok = (size_t)BWIN * NWIN;           // 401408 token rows
    _Float16* qkv_ws  = (_Float16*)d_ws;              // 3 * tok * 256 halves
    _Float16* attn_ws = qkv_ws + (size_t)3 * tok * DIMC;

    dim3 blk(128);
    qkv_gemm_kernel<<<dim3((unsigned)(tok / 64), 12), blk, 0, stream>>>(x, Wqkv, bqkv, qkv_ws);
    attn_kernel<<<dim3(BWIN * HEADS), blk, 0, stream>>>(qkv_ws, mask, attn_ws);
    proj_gemm_kernel<<<dim3((unsigned)(tok / 64), 4), blk, 0, stream>>>(attn_ws, Wproj, bproj, out);
}
